// ATSA_56384330662502
// MI455X (gfx1250) — matrix operand order repaired and checked by reading, not compiled, not run on hardware
//
#include <hip/hip_runtime.h>
#include <hip/hip_bf16.h>
#include <math.h>
#include <float.h>

// ---------------------------------------------------------------------------
// Shapes (fixed by the reference): B=64, N=2048, C=1024, H=512, K=20 (pad 32)
// ---------------------------------------------------------------------------
#define B_    64
#define N_    2048
#define C_    1024
#define H_    512
#define KMAX  20
#define KPAD  32

typedef __attribute__((ext_vector_type(16))) __bf16 v16bf;
typedef __attribute__((ext_vector_type(8)))  __bf16 v8bf;
typedef __attribute__((ext_vector_type(8)))  float  v8f;

__device__ __forceinline__ unsigned short f2bfu(float f) {
  unsigned u = __builtin_bit_cast(unsigned, f);
  u += 0x7FFFu + ((u >> 16) & 1u);               // round-to-nearest-even
  return (unsigned short)(u >> 16);
}
__device__ __forceinline__ __bf16 f2bf(float f) {
  unsigned short s = f2bfu(f);
  return __builtin_bit_cast(__bf16, s);
}
__device__ __forceinline__ v8f vzero8() {
  v8f z = {0.f, 0.f, 0.f, 0.f, 0.f, 0.f, 0.f, 0.f};
  return z;
}
__device__ __forceinline__ v16bf pack16(v8bf lo, v8bf hi) {
  v16bf r;
#pragma unroll
  for (int i = 0; i < 8; ++i) { r[i] = lo[i]; r[i + 8] = hi[i]; }
  return r;
}
__device__ __forceinline__ v8f wmma_bf16(v16bf a, v16bf b, v8f c) {
  return __builtin_amdgcn_wmma_f32_16x16x32_bf16(false, a, false, b,
                                                 (short)0, c, false, false);
}

// ---------------------------------------------------------------------------
// Kernel P: swizzle a [1024,512] fp32 weight into WMMA-B bf16 fragments.
// Fragment (ks,ht) = B[k=ks*32..+31, n=ht*16..+15].
// Layout per fragment (512 halves): two 256-half planes; within a plane:
// lane*8 + ee, with element e = plane*8+ee holding B[kbase + plane*16 + (lane>=16?8:0) + ee,
// ht*16 + (lane&15)].  Each wave lane then loads its 16 halves as two b128s.
// ---------------------------------------------------------------------------
__global__ void __launch_bounds__(512)
prep_frag(const float* __restrict__ pw1, const float* __restrict__ rw1,
          __bf16* __restrict__ pf, __bf16* __restrict__ rf) {
  const float* src = blockIdx.y ? rw1 : pw1;
  __bf16* dst      = blockIdx.y ? rf  : pf;
  int t    = threadIdx.x;            // 0..511
  int frag = blockIdx.x;             // ks*32 + ht, 1024 frags
  int ks = frag >> 5, ht = frag & 31;
  int p    = t >> 8;                 // plane
  int lane = (t >> 3) & 31;
  int ee   = t & 7;
  int e    = p * 8 + ee;
  int kl   = p * 16 + ((lane >= 16) ? 8 : 0) + ee;
  int n    = lane & 15;
  float v  = src[(ks * 32 + kl) * H_ + ht * 16 + n];
  dst[frag * 512 + t] = f2bf(v);
}

// ---------------------------------------------------------------------------
// Kernel 1 (the big one): one pass over tokens.
//   - stages 64 rows x 1024 cols fp32 -> bf16 in LDS (stride padded: 1032 halves
//     = 516 dwords = 4 mod 64 banks -> conflict-free 16B ds loads)
//   - fp32 column sums -> atomicAdd into total[b, c]
//   - 8 waves, 2x2 register tiling: wave w: m-pair = w&1 (32 rows),
//     h-octant = w>>1 (8 of 32 h-tiles, processed in pairs).
//     4 independent accumulators -> 2 memory instructions per WMMA.
//     Epilogue fuses +p_b1, relu, *p_w2 into running per-row scalars.
//   - shfl_xor reduce over the 16 N-lanes, atomicAdd partial into imp[b, n].
// ---------------------------------------------------------------------------
#define K1_ROWS   64
#define K1_STRIDE 1032  /* halves */

__global__ void __launch_bounds__(256)
k1_imp_total(const float* __restrict__ tokens,
             const __bf16* __restrict__ pf,
             const float* __restrict__ p_b1,
             const float* __restrict__ p_w2,
             float* __restrict__ total,
             float* __restrict__ imp) {
  extern __shared__ __bf16 lds[];  // 64 * 1032 halves = 132096 bytes
  const int t    = threadIdx.x;
  const int b    = blockIdx.x >> 5;            // 32 WGs per batch
  const int row0 = (blockIdx.x & 31) * K1_ROWS;

  // ---- stage tokens -> LDS bf16, accumulate fp32 column sums ----
  const float4* tok4 = (const float4*)tokens;
  float s0 = 0.f, s1 = 0.f, s2 = 0.f, s3 = 0.f;
  const int colbase = t * 4;                   // 256 threads * 4 = 1024 cols
  for (int r = 0; r < K1_ROWS; ++r) {
    float4 v = tok4[(size_t)(b * N_ + row0 + r) * (C_ / 4) + t];
    s0 += v.x; s1 += v.y; s2 += v.z; s3 += v.w;
    unsigned lo = (unsigned)f2bfu(v.x) | ((unsigned)f2bfu(v.y) << 16);
    unsigned hi = (unsigned)f2bfu(v.z) | ((unsigned)f2bfu(v.w) << 16);
    uint2 pk; pk.x = lo; pk.y = hi;
    *(uint2*)(lds + r * K1_STRIDE + colbase) = pk;
  }
  float* tb = total + b * C_ + colbase;
  atomicAdd(tb + 0, s0); atomicAdd(tb + 1, s1);
  atomicAdd(tb + 2, s2); atomicAdd(tb + 3, s3);
  __syncthreads();

  // ---- WMMA compute, 2 m-tiles x 2 h-tiles per wave ----
  const int w     = t >> 5;
  const int lane  = t & 31;
  const int mpair = w & 1;       // m_tiles {2*mpair, 2*mpair+1}
  const int hoct  = w >> 1;      // h_tiles [hoct*8, hoct*8+8)
  const int lhalf = lane >> 4;   // A lanes 0-15: K 0..7/16..23; 16-31: 8..15/24..31
  const __bf16* arow0 = lds + ((mpair * 2) * 16 + (lane & 15)) * K1_STRIDE;
  const __bf16* arow1 = arow0 + 16 * K1_STRIDE;

  v8f simp0 = vzero8(), simp1 = vzero8();
  for (int hp = 0; hp < 4; ++hp) {
    const int ht0 = hoct * 8 + hp * 2;
    v8f acc00 = vzero8(), acc01 = vzero8();
    v8f acc10 = vzero8(), acc11 = vzero8();
#pragma unroll 2
    for (int ks = 0; ks < 32; ++ks) {
      const int koff = ks * 32 + lhalf * 8;
      v16bf a0 = pack16(*(const v8bf*)(arow0 + koff),
                        *(const v8bf*)(arow0 + koff + 16));
      v16bf a1 = pack16(*(const v8bf*)(arow1 + koff),
                        *(const v8bf*)(arow1 + koff + 16));
      const __bf16* fb0 = pf + (ks * 32 + ht0) * 512 + lane * 8;
      v16bf b0 = pack16(*(const v8bf*)(fb0),       *(const v8bf*)(fb0 + 256));
      v16bf b1 = pack16(*(const v8bf*)(fb0 + 512), *(const v8bf*)(fb0 + 768));
      acc00 = wmma_bf16(a0, b0, acc00);
      acc01 = wmma_bf16(a0, b1, acc01);
      acc10 = wmma_bf16(a1, b0, acc10);
      acc11 = wmma_bf16(a1, b1, acc11);
    }
    {
      const int h0 = ht0 * 16 + (lane & 15);
      const float bias0 = p_b1[h0], w20 = p_w2[h0];
      const int h1 = h0 + 16;
      const float bias1 = p_b1[h1], w21 = p_w2[h1];
#pragma unroll
      for (int r = 0; r < 8; ++r) {
        simp0[r] = fmaf(fmaxf(acc00[r] + bias0, 0.f), w20, simp0[r]);
        simp1[r] = fmaf(fmaxf(acc10[r] + bias0, 0.f), w20, simp1[r]);
        simp0[r] = fmaf(fmaxf(acc01[r] + bias1, 0.f), w21, simp0[r]);
        simp1[r] = fmaf(fmaxf(acc11[r] + bias1, 0.f), w21, simp1[r]);
      }
    }
  }
  // reduce over the 16 N-lanes (lanes 0-15 hold rows 0-7, 16-31 hold rows 8-15)
#pragma unroll
  for (int m = 1; m <= 8; m <<= 1) {
#pragma unroll
    for (int r = 0; r < 8; ++r) {
      simp0[r] += __shfl_xor(simp0[r], m, 32);
      simp1[r] += __shfl_xor(simp1[r], m, 32);
    }
  }
  if ((lane & 15) == 0) {
    const int base0 = b * N_ + row0 + (mpair * 2) * 16 + lhalf * 8;
#pragma unroll
    for (int r = 0; r < 8; ++r) atomicAdd(imp + base0 + r, simp0[r]);
#pragma unroll
    for (int r = 0; r < 8; ++r) atomicAdd(imp + base0 + 16 + r, simp1[r]);
  }
}

// ---------------------------------------------------------------------------
// Router: feat = relu(mean @ enc_w + enc_b); alpha, k heads; k / a_k ints.
// ---------------------------------------------------------------------------
__global__ void __launch_bounds__(512)
router_kernel(const float* __restrict__ total,
              const float* __restrict__ enc_w, const float* __restrict__ enc_b,
              const float* __restrict__ a_w1,  const float* __restrict__ a_b1,
              const float* __restrict__ a_w2,  const float* __restrict__ a_b2,
              const float* __restrict__ k_w1,  const float* __restrict__ k_b1,
              const float* __restrict__ k_w2,  const float* __restrict__ k_b2,
              int* __restrict__ kkb, int* __restrict__ akb) {
  __shared__ float s_mean[C_];
  __shared__ float s_feat[H_];
  __shared__ float s_red[512];
  const int b = blockIdx.x, t = threadIdx.x;
  s_mean[t]       = total[b * C_ + t]       * (1.f / (float)N_);
  s_mean[t + 512] = total[b * C_ + 512 + t] * (1.f / (float)N_);
  __syncthreads();
  {
    float acc = enc_b[t];
    for (int c = 0; c < C_; ++c) acc = fmaf(s_mean[c], enc_w[c * H_ + t], acc);
    s_feat[t] = fmaxf(acc, 0.f);
  }
  __syncthreads();
  {
    const int j = t & 255;
    const bool isK = (t >= 256);
    const float* w1 = isK ? k_w1 : a_w1;
    const float* b1 = isK ? k_b1 : a_b1;
    const float* w2 = isK ? k_w2 : a_w2;
    float h = b1[j];
    for (int x = 0; x < H_; ++x) h = fmaf(s_feat[x], w1[x * 256 + j], h);
    h = fmaxf(h, 0.f);
    s_red[t] = h * w2[j];
  }
  __syncthreads();
  for (int s = 128; s > 0; s >>= 1) {
    if ((t & 255) < s) s_red[t] += s_red[t + s];
    __syncthreads();
  }
  if (t == 0) {
    float al = s_red[0]   + a_b2[0];
    float kl = s_red[256] + k_b2[0];
    float alpha = 1.f / (1.f + expf(-al));
    float kraw  = (kl > 0.f) ? (kl + log1pf(expf(-kl))) : log1pf(expf(kl));
    int kk = (int)rintf(kraw);
    if (kk < 1) kk = 1;
    if (kk > KMAX) kk = KMAX;
    int ak = (int)floorf(alpha * (float)kk);
    if (ak < 1) ak = 1;
    if (ak > kk) ak = kk;
    kkb[b] = kk; akb[b] = ak;
  }
}

// ---------------------------------------------------------------------------
// Top-K: iterative argmax (softmax skipped — monotonic; p_b2 rank-invariant).
// ---------------------------------------------------------------------------
__global__ void __launch_bounds__(256)
topk_kernel(const float* __restrict__ imp, int* __restrict__ idxb) {
  __shared__ float sv[N_];
  __shared__ float rv[256];
  __shared__ int   ri[256];
  const int b = blockIdx.x, t = threadIdx.x;
  for (int i = t; i < N_; i += 256) sv[i] = imp[b * N_ + i];
  __syncthreads();
  for (int it = 0; it < KMAX; ++it) {
    float bv = -FLT_MAX; int bi = 0x7fffffff;
    for (int i = t; i < N_; i += 256) {
      float v = sv[i];
      if (v > bv) { bv = v; bi = i; }
    }
    rv[t] = bv; ri[t] = bi;
    __syncthreads();
    for (int s = 128; s > 0; s >>= 1) {
      if (t < s) {
        if (rv[t + s] > rv[t] || (rv[t + s] == rv[t] && ri[t + s] < ri[t])) {
          rv[t] = rv[t + s]; ri[t] = ri[t + s];
        }
      }
      __syncthreads();
    }
    if (t == 0) { idxb[b * KPAD + it] = ri[0]; sv[ri[0]] = -FLT_MAX; }
    __syncthreads();
  }
}

// ---------------------------------------------------------------------------
// Gather: top-K rows -> bf16 (pad to 32 rows/batch with zeros) + masked sums.
// ---------------------------------------------------------------------------
__global__ void __launch_bounds__(256)
gather_kernel(const float* __restrict__ tokens, const int* __restrict__ idxb,
              const int* __restrict__ kkb, const int* __restrict__ akb,
              __bf16* __restrict__ topkbf,
              float* __restrict__ stopk, float* __restrict__ srem) {
  const int b = blockIdx.x, t = threadIdx.x;
  const int kk = kkb[b], ak = akb[b];
  float s0 = 0, s1 = 0, s2 = 0, s3 = 0;
  float r0 = 0, r1 = 0, r2 = 0, r3 = 0;
  const float4* tok4 = (const float4*)tokens;
  for (int i = 0; i < KPAD; ++i) {
    unsigned short h0 = 0, h1 = 0, h2 = 0, h3 = 0;
    if (i < KMAX) {
      const int row = idxb[b * KPAD + i];
      float4 v = tok4[(size_t)(b * N_ + row) * (C_ / 4) + t];
      if (i < kk) {
        s0 += v.x; s1 += v.y; s2 += v.z; s3 += v.w;
        if (i >= ak) { r0 += v.x; r1 += v.y; r2 += v.z; r3 += v.w; }
      }
      h0 = f2bfu(v.x); h1 = f2bfu(v.y); h2 = f2bfu(v.z); h3 = f2bfu(v.w);
    }
    unsigned lo = (unsigned)h0 | ((unsigned)h1 << 16);
    unsigned hi = (unsigned)h2 | ((unsigned)h3 << 16);
    uint2 pk; pk.x = lo; pk.y = hi;
    *(uint2*)(topkbf + (size_t)(b * KPAD + i) * C_ + t * 4) = pk;
  }
  const int c = b * C_ + t * 4;
  stopk[c + 0] = s0; stopk[c + 1] = s1; stopk[c + 2] = s2; stopk[c + 3] = s3;
  srem[c + 0]  = r0; srem[c + 1]  = r1; srem[c + 2]  = r2; srem[c + 3]  = r3;
}

// ---------------------------------------------------------------------------
// G2: refiner hidden via WMMA (2x2 tiling); masked rows (i < a_k) -> hsum.
// ---------------------------------------------------------------------------
__global__ void __launch_bounds__(256)
g2_hidden(const __bf16* __restrict__ topkbf, const __bf16* __restrict__ rf,
          const float* __restrict__ r_b1, const int* __restrict__ akb,
          float* __restrict__ hsum) {
  const int t = threadIdx.x;
  const int w = t >> 5, lane = t & 31;
  const int mt0 = (blockIdx.x * 8 + w) * 2;   // 8 blocks x 8 waves x 2 = 128 m-tiles
  const int lhalf = lane >> 4;
  const __bf16* arow0 = topkbf + (size_t)(mt0 * 16 + (lane & 15)) * C_;
  const __bf16* arow1 = arow0 + 16 * C_;
  for (int hp = 0; hp < 16; ++hp) {
    const int ht0 = hp * 2;
    v8f acc00 = vzero8(), acc01 = vzero8();
    v8f acc10 = vzero8(), acc11 = vzero8();
#pragma unroll 2
    for (int ks = 0; ks < 32; ++ks) {
      const int koff = ks * 32 + lhalf * 8;
      v16bf a0 = pack16(*(const v8bf*)(arow0 + koff),
                        *(const v8bf*)(arow0 + koff + 16));
      v16bf a1 = pack16(*(const v8bf*)(arow1 + koff),
                        *(const v8bf*)(arow1 + koff + 16));
      const __bf16* fb0 = rf + (ks * 32 + ht0) * 512 + lane * 8;
      v16bf b0 = pack16(*(const v8bf*)(fb0),       *(const v8bf*)(fb0 + 256));
      v16bf b1 = pack16(*(const v8bf*)(fb0 + 512), *(const v8bf*)(fb0 + 768));
      acc00 = wmma_bf16(a0, b0, acc00);
      acc01 = wmma_bf16(a0, b1, acc01);
      acc10 = wmma_bf16(a1, b0, acc10);
      acc11 = wmma_bf16(a1, b1, acc11);
    }
    const int h0 = ht0 * 16 + (lane & 15);
    const int h1 = h0 + 16;
    const float bias0 = r_b1[h0], bias1 = r_b1[h1];
#pragma unroll
    for (int r = 0; r < 8; ++r) {
      const int rm0 = mt0 * 16 + r + lhalf * 8;
      const int rm1 = rm0 + 16;
      const int bb0 = rm0 >> 5, ii0 = rm0 & 31;   // KPAD == 32
      const int bb1 = rm1 >> 5, ii1 = rm1 & 31;
      float v00 = fmaxf(acc00[r] + bias0, 0.f);
      float v01 = fmaxf(acc01[r] + bias1, 0.f);
      float v10 = fmaxf(acc10[r] + bias0, 0.f);
      float v11 = fmaxf(acc11[r] + bias1, 0.f);
      if (ii0 < akb[bb0]) {
        atomicAdd(hsum + bb0 * H_ + h0, v00);
        atomicAdd(hsum + bb0 * H_ + h1, v01);
      }
      if (ii1 < akb[bb1]) {
        atomicAdd(hsum + bb1 * H_ + h0, v10);
        atomicAdd(hsum + bb1 * H_ + h1, v11);
      }
    }
  }
}

// ---------------------------------------------------------------------------
// Final: sum_ref = hsum @ r_w2 + a_k*r_b2; pooled; agg; final MLP -> out.
// ---------------------------------------------------------------------------
__global__ void __launch_bounds__(1024)
final_kernel(const float* __restrict__ hsum, const float* __restrict__ total,
             const float* __restrict__ stopk, const float* __restrict__ srem,
             const int* __restrict__ akb,
             const float* __restrict__ r_w2, const float* __restrict__ r_b2,
             const float* __restrict__ f_w1, const float* __restrict__ f_b1,
             const float* __restrict__ f_w2, const float* __restrict__ f_b2,
             float* __restrict__ out) {
  __shared__ float s_hsum[H_];
  __shared__ float s_agg[C_];
  __shared__ float s_fhid[H_];
  const int b = blockIdx.x, t = threadIdx.x;
  if (t < H_) s_hsum[t] = hsum[b * H_ + t];
  __syncthreads();
  const float akf = (float)akb[b];
  {
    float sref = akf * r_b2[t];
    for (int h = 0; h < H_; ++h) sref = fmaf(s_hsum[h], r_w2[h * C_ + t], sref);
    const int c = b * C_ + t;
    float pooled = (srem[c] + total[c] - stopk[c]) / ((float)N_ - akf);
    s_agg[t] = (sref + pooled) / (akf + 1.f);
  }
  __syncthreads();
  if (t < H_) {
    float a = f_b1[t];
    for (int c = 0; c < C_; ++c) a = fmaf(s_agg[c], f_w1[c * H_ + t], a);
    s_fhid[t] = fmaxf(a, 0.f);
  }
  __syncthreads();
  float o = f_b2[t];
  for (int h = 0; h < H_; ++h) o = fmaf(s_fhid[h], f_w2[h * C_ + t], o);
  out[b * C_ + t] = o;
}

// ---------------------------------------------------------------------------
// Host launcher
// ---------------------------------------------------------------------------
extern "C" void kernel_launch(void* const* d_in, const int* in_sizes, int n_in,
                              void* d_out, int out_size, void* d_ws, size_t ws_size,
                              hipStream_t stream) {
  (void)in_sizes; (void)n_in; (void)out_size; (void)ws_size;
  const float* tokens = (const float*)d_in[0];
  const float* enc_w  = (const float*)d_in[1];
  const float* enc_b  = (const float*)d_in[2];
  const float* a_w1   = (const float*)d_in[3];
  const float* a_b1   = (const float*)d_in[4];
  const float* a_w2   = (const float*)d_in[5];
  const float* a_b2   = (const float*)d_in[6];
  const float* k_w1   = (const float*)d_in[7];
  const float* k_b1   = (const float*)d_in[8];
  const float* k_w2   = (const float*)d_in[9];
  const float* k_b2   = (const float*)d_in[10];
  const float* p_w1   = (const float*)d_in[11];
  const float* p_b1   = (const float*)d_in[12];
  const float* p_w2   = (const float*)d_in[13];
  const float* r_w1   = (const float*)d_in[15];
  const float* r_b1   = (const float*)d_in[16];
  const float* r_w2   = (const float*)d_in[17];
  const float* r_b2   = (const float*)d_in[18];
  const float* f_w1   = (const float*)d_in[19];
  const float* f_b1   = (const float*)d_in[20];
  const float* f_w2   = (const float*)d_in[21];
  const float* f_b2   = (const float*)d_in[22];
  float* outp = (float*)d_out;

  char* ws = (char*)d_ws;
  // workspace layout (bytes)
  float*  w_total = (float*)(ws + 0);          //  64*1024 f  (zeroed)
  float*  w_imp   = (float*)(ws + 262144);     //  64*2048 f  (zeroed)
  float*  w_hsum  = (float*)(ws + 786432);     //  64*512  f  (zeroed)
  float*  w_stopk = (float*)(ws + 917504);     //  64*1024 f
  float*  w_srem  = (float*)(ws + 1179648);    //  64*1024 f
  int*    w_kk    = (int*)  (ws + 1441792);    //  64 i
  int*    w_ak    = (int*)  (ws + 1442304);    //  64 i
  int*    w_idx   = (int*)  (ws + 1442816);    //  64*32 i
  __bf16* w_pw1f  = (__bf16*)(ws + 1451008);   //  1024*512 bf16 fragments
  __bf16* w_rw1f  = (__bf16*)(ws + 2499584);   //  1024*512 bf16 fragments
  __bf16* w_topk  = (__bf16*)(ws + 3548160);   //  64*32*1024 bf16

  // zero the accumulation buffers (total, imp, hsum are contiguous at 0)
  hipMemsetAsync(d_ws, 0, 917504, stream);

  // 1) swizzle p_w1 / r_w1 into WMMA B-fragment layout
  prep_frag<<<dim3(1024, 2), 512, 0, stream>>>(p_w1, r_w1, w_pw1f, w_rw1f);

  // 2) single pass over tokens: column sums + fused priority MLP -> imp
  const int k1_lds = K1_ROWS * K1_STRIDE * 2;  // 132096 B (<320KB LDS/WGP)
  hipFuncSetAttribute((const void*)k1_imp_total,
                      hipFuncAttributeMaxDynamicSharedMemorySize, k1_lds);
  k1_imp_total<<<(B_ * N_) / K1_ROWS, 256, k1_lds, stream>>>(
      tokens, w_pw1f, p_b1, p_w2, w_total, w_imp);

  // 3) router heads -> k, a_k
  router_kernel<<<B_, 512, 0, stream>>>(w_total, enc_w, enc_b, a_w1, a_b1,
                                        a_w2, a_b2, k_w1, k_b1, k_w2, k_b2,
                                        w_kk, w_ak);

  // 4) top-K indices
  topk_kernel<<<B_, 256, 0, stream>>>(w_imp, w_idx);

  // 5) gather top-K rows (bf16, zero-padded) + masked sums
  gather_kernel<<<B_, 256, 0, stream>>>(tokens, w_idx, w_kk, w_ak,
                                        w_topk, w_stopk, w_srem);

  // 6) refiner hidden (WMMA, 2x2 tiling) -> masked row-sum hsum
  g2_hidden<<<8, 256, 0, stream>>>(w_topk, w_rw1f, r_b1, w_ak, w_hsum);

  // 7) collapse refiner layer 2, pooled, agg, final MLP -> out
  final_kernel<<<B_, 1024, 0, stream>>>(w_hsum, w_total, w_stopk, w_srem,
                                        w_ak, r_w2, r_b2, f_w1, f_b1,
                                        f_w2, f_b2, outp);
}
